// lstm_seq2seq_69088843924350
// MI455X (gfx1250) — matrix operand order repaired and checked by reading, not compiled, not run on hardware
//
#include <hip/hip_runtime.h>
#include <hip/hip_bf16.h>
#include <cstdint>
#include <cstddef>

typedef __attribute__((ext_vector_type(16))) _Float16 v16h;
typedef __attribute__((ext_vector_type(8)))  _Float16 v8h;
typedef __attribute__((ext_vector_type(8)))  float    v8f;

#define DEV __device__ __forceinline__

// ---- problem dims (from setup_inputs) ----
constexpr int B_   = 512;
constexpr int T_   = 256;
constexpr int DIN  = 32;
constexpr int H_   = 128;
constexpr int G4   = 4 * H_;          // 512 gate columns
constexpr int BT   = 32;              // batch rows per workgroup
constexpr int NWG  = B_ / BT;         // 16 workgroups
constexpr int LDH  = H_ + 8;          // padded LDS row stride (halves)
constexpr int LDX  = DIN + 8;
constexpr int KF0  = (DIN + H_) / 32; // 5  (layer0 K=160)
constexpr int KF1  = (H_ + H_) / 32;  // 8  (layer1 K=256)
constexpr int KFL  = H_ / 32;         // 4  (linear K=128)

// ---- fast activations: single v_exp_f32 + single v_rcp_f32, no IEEE div ----
DEV float rcp_(float x)  { return __builtin_amdgcn_rcpf(x); }
DEV float sig_(float x)  { return rcp_(1.0f + __expf(-x)); }
DEV float tanh_(float x) { return 1.0f - 2.0f * rcp_(1.0f + __expf(2.0f * x)); }

DEV v8f splat8(float s) {
  v8f v;
  #pragma unroll
  for (int e = 0; e < 8; ++e) v[e] = s;
  return v;
}

DEV v8f wmma16(v16h a, v16h b, v8f c) {
  // v_wmma_f32_16x16x32_f16: D = A(16x32 f16) * B(32x16 f16) + C(16x16 f32)
  return __builtin_amdgcn_wmma_f32_16x16x32_f16(false, a, false, b, (short)0, c, false, false);
}

// A-fragment (16x32 f16, M=lane&15; K = kf*32 + (lane>=16?8:0) + {0..7, 16..23})
DEV v16h a_lds(const _Float16* row, int kf, int hi16) {
  const int k0 = kf * 32 + hi16 * 8;
  v8h lo = *(const v8h*)(row + k0);
  v8h hi = *(const v8h*)(row + k0 + 16);
  return __builtin_shufflevector(lo, hi, 0,1,2,3,4,5,6,7,8,9,10,11,12,13,14,15);
}

// A-fragment from global f32 row (K=32 only: kf==0), convert to f16
DEV v16h a_gf32(const float* row, int hi16) {
  const int k0 = hi16 * 8;
  float4 x0 = *(const float4*)(row + k0);
  float4 x1 = *(const float4*)(row + k0 + 4);
  float4 y0 = *(const float4*)(row + k0 + 16);
  float4 y1 = *(const float4*)(row + k0 + 20);
  v16h r;
  r[0]=(_Float16)x0.x; r[1]=(_Float16)x0.y; r[2]=(_Float16)x0.z; r[3]=(_Float16)x0.w;
  r[4]=(_Float16)x1.x; r[5]=(_Float16)x1.y; r[6]=(_Float16)x1.z; r[7]=(_Float16)x1.w;
  r[8]=(_Float16)y0.x; r[9]=(_Float16)y0.y; r[10]=(_Float16)y0.z; r[11]=(_Float16)y0.w;
  r[12]=(_Float16)y1.x; r[13]=(_Float16)y1.y; r[14]=(_Float16)y1.z; r[15]=(_Float16)y1.w;
  return r;
}

// B-fragment (32x16 f16) pre-arranged: element j of lane L at contiguous addr
DEV v16h b_frag(const _Float16* wf, int tile, int lane) {
  return *(const v16h*)(wf + ((size_t)tile * 32 + lane) * 16);
}

// ---- prep: convert W_cat=[Wih|Whh] (f32) into f16 B-fragment layout ----
// element j of (tile=nt*KF+kf, lane): W_cat[n = nt*16+(lane&15)][k = kf*32+(lane>=16?8:0)+(j<8?j:j+8)]
__global__ void prep_wfrag(_Float16* __restrict__ dst, const float* __restrict__ Wih,
                           const float* __restrict__ Whh, int Kin, int KF) {
  const int tile = blockIdx.x, lane = threadIdx.x;
  const int nt = tile / KF, kf = tile % KF;
  const int n  = nt * 16 + (lane & 15);
  const int kb = kf * 32 + ((lane >= 16) ? 8 : 0);
  _Float16* o = dst + ((size_t)tile * 32 + lane) * 16;
  #pragma unroll
  for (int j = 0; j < 16; ++j) {
    const int k = kb + (j < 8 ? j : j + 8);
    const float v = (k < Kin) ? Wih[(size_t)n * Kin + k] : Whh[(size_t)n * H_ + (k - Kin)];
    o[j] = (_Float16)v;
  }
}

__global__ void bias_add(float* __restrict__ dst, const float* __restrict__ a,
                         const float* __restrict__ b, int n) {
  const int i = blockIdx.x * blockDim.x + threadIdx.x;
  if (i < n) dst[i] = a[i] + b[i];
}

// =====================================================================
// Encoder: both LSTM layers fused, full T=256 time loop in one kernel.
// 16 workgroups x 256 threads (8 waves). Wave w owns h columns [w*16,w*16+16).
// Layer-1 weights (256KB f16) cached in LDS; h0/h1 state in LDS f16.
// =====================================================================
__global__ __launch_bounds__(256) void enc_kernel(
    const float* __restrict__ xg,
    const _Float16* __restrict__ wf0, const float* __restrict__ bs0,
    const _Float16* __restrict__ wf1, const float* __restrict__ bs1,
    float* __restrict__ eh0, float* __restrict__ ec0,
    float* __restrict__ eh1, float* __restrict__ ec1) {
  extern __shared__ _Float16 smem[];
  _Float16* WL1 = smem;                       // 512*256 halves = 256 KB
  _Float16* Hs0 = smem + G4 * (H_ + H_);      // BT x LDH
  _Float16* Hs1 = Hs0 + BT * LDH;

  const int tid = threadIdx.x;
  const int lane = tid & 31, w = tid >> 5;
  const int hi16 = lane >> 4, l15 = lane & 15;
  const int m0 = blockIdx.x * BT;
  const int nw = w;                           // h n-tile owned by this wave

  { // stage layer-1 weight fragments L2 -> LDS once
    const uint4* src = (const uint4*)wf1;
    uint4* dst = (uint4*)WL1;
    for (int i = tid; i < G4 * (H_ + H_) * 2 / 16; i += 256) dst[i] = src[i];
  }
  for (int i = tid; i < BT * LDH; i += 256) { Hs0[i] = (_Float16)0.0f; Hs1[i] = (_Float16)0.0f; }

  float b0v[4], b1v[4];
  #pragma unroll
  for (int g = 0; g < 4; ++g) {
    b0v[g] = bs0[g * H_ + nw * 16 + l15];
    b1v[g] = bs1[g * H_ + nw * 16 + l15];
  }
  v8f c0[2], c1[2];
  #pragma unroll
  for (int mt = 0; mt < 2; ++mt) { c0[mt] = splat8(0.f); c1[mt] = splat8(0.f); }
  __syncthreads();

  for (int t = 0; t < T_; ++t) {
    // -------- layer 0: gates = [x_t | h0] @ W0^T + b0
    v8f acc[2][4];
    #pragma unroll
    for (int mt = 0; mt < 2; ++mt)
      #pragma unroll
      for (int g = 0; g < 4; ++g) acc[mt][g] = splat8(b0v[g]);

    #pragma unroll
    for (int mt = 0; mt < 2; ++mt) {   // K-frag 0: x_t from global
      const float* row = xg + ((size_t)(m0 + mt * 16 + l15) * T_ + t) * DIN;
      v16h a = a_gf32(row, hi16);
      #pragma unroll
      for (int g = 0; g < 4; ++g)
        acc[mt][g] = wmma16(a, b_frag(wf0, (g * 8 + nw) * KF0 + 0, lane), acc[mt][g]);
    }
    #pragma unroll
    for (int kf = 1; kf < KF0; ++kf) { // K-frags 1..4: h0 from LDS
      v16h a[2];
      #pragma unroll
      for (int mt = 0; mt < 2; ++mt) a[mt] = a_lds(Hs0 + (mt * 16 + l15) * LDH, kf - 1, hi16);
      #pragma unroll
      for (int g = 0; g < 4; ++g) {
        v16h b = b_frag(wf0, (g * 8 + nw) * KF0 + kf, lane);
        #pragma unroll
        for (int mt = 0; mt < 2; ++mt) acc[mt][g] = wmma16(a[mt], b, acc[mt][g]);
      }
    }
    v8f h0n[2];
    #pragma unroll
    for (int mt = 0; mt < 2; ++mt)
      #pragma unroll
      for (int e = 0; e < 8; ++e) {
        float cc = sig_(acc[mt][1][e]) * c0[mt][e] + sig_(acc[mt][0][e]) * tanh_(acc[mt][2][e]);
        c0[mt][e] = cc;
        h0n[mt][e] = sig_(acc[mt][3][e]) * tanh_(cc);
      }
    __syncthreads();
    #pragma unroll
    for (int mt = 0; mt < 2; ++mt)
      #pragma unroll
      for (int e = 0; e < 8; ++e)
        Hs0[(mt * 16 + e + hi16 * 8) * LDH + nw * 16 + l15] = (_Float16)h0n[mt][e];
    __syncthreads();

    // -------- layer 1: gates = [h0_t | h1] @ W1^T + b1 (weights from LDS)
    #pragma unroll
    for (int mt = 0; mt < 2; ++mt)
      #pragma unroll
      for (int g = 0; g < 4; ++g) acc[mt][g] = splat8(b1v[g]);
    #pragma unroll
    for (int kf = 0; kf < KF1; ++kf) {
      const _Float16* src = (kf < 4) ? Hs0 : Hs1;
      const int kl = (kf < 4) ? kf : kf - 4;
      v16h a[2];
      #pragma unroll
      for (int mt = 0; mt < 2; ++mt) a[mt] = a_lds(src + (mt * 16 + l15) * LDH, kl, hi16);
      #pragma unroll
      for (int g = 0; g < 4; ++g) {
        v16h b = b_frag(WL1, (g * 8 + nw) * KF1 + kf, lane);
        #pragma unroll
        for (int mt = 0; mt < 2; ++mt) acc[mt][g] = wmma16(a[mt], b, acc[mt][g]);
      }
    }
    v8f h1n[2];
    #pragma unroll
    for (int mt = 0; mt < 2; ++mt)
      #pragma unroll
      for (int e = 0; e < 8; ++e) {
        float cc = sig_(acc[mt][1][e]) * c1[mt][e] + sig_(acc[mt][0][e]) * tanh_(acc[mt][2][e]);
        c1[mt][e] = cc;
        h1n[mt][e] = sig_(acc[mt][3][e]) * tanh_(cc);
      }
    __syncthreads();
    #pragma unroll
    for (int mt = 0; mt < 2; ++mt)
      #pragma unroll
      for (int e = 0; e < 8; ++e)
        Hs1[(mt * 16 + e + hi16 * 8) * LDH + nw * 16 + l15] = (_Float16)h1n[mt][e];
    __syncthreads();
  }

  // final states for the decoder
  #pragma unroll
  for (int mt = 0; mt < 2; ++mt)
    #pragma unroll
    for (int e = 0; e < 8; ++e) {
      const int row = m0 + mt * 16 + e + hi16 * 8;
      const int col = nw * 16 + l15;
      ec0[(size_t)row * H_ + col] = c0[mt][e];
      ec1[(size_t)row * H_ + col] = c1[mt][e];
    }
  for (int i = tid; i < BT * H_; i += 256) {
    const int r = i / H_, cix = i % H_;
    eh0[(size_t)(m0 + r) * H_ + cix] = (float)Hs0[r * LDH + cix];
    eh1[(size_t)(m0 + r) * H_ + cix] = (float)Hs1[r * LDH + cix];
  }
}

// =====================================================================
// Decoder: 2 LSTM cells + 128->32 linear with autoregressive feedback,
// full TGT loop in one kernel. Same tiling as the encoder.
// =====================================================================
__global__ __launch_bounds__(256) void dec_kernel(
    const float* __restrict__ xg, const int* __restrict__ tgt,
    const _Float16* __restrict__ wf0, const float* __restrict__ bs0,
    const _Float16* __restrict__ wf1, const float* __restrict__ bs1,
    const _Float16* __restrict__ wfl, const float* __restrict__ bl,
    const float* __restrict__ eh0, const float* __restrict__ ec0,
    const float* __restrict__ eh1, const float* __restrict__ ec1,
    float* __restrict__ out) {
  extern __shared__ _Float16 smem[];
  _Float16* WL1 = smem;
  _Float16* Hs0 = smem + G4 * (H_ + H_);
  _Float16* Hs1 = Hs0 + BT * LDH;
  _Float16* Xs  = Hs1 + BT * LDH;            // BT x LDX feedback input

  const int tid = threadIdx.x;
  const int lane = tid & 31, w = tid >> 5;
  const int hi16 = lane >> 4, l15 = lane & 15;
  const int m0 = blockIdx.x * BT;
  const int nw = w;
  const int tlen = tgt[0];

  {
    const uint4* src = (const uint4*)wf1;
    uint4* dst = (uint4*)WL1;
    for (int i = tid; i < G4 * (H_ + H_) * 2 / 16; i += 256) dst[i] = src[i];
  }
  for (int i = tid; i < BT * H_; i += 256) {
    const int r = i / H_, cix = i % H_;
    Hs0[r * LDH + cix] = (_Float16)eh0[(size_t)(m0 + r) * H_ + cix];
    Hs1[r * LDH + cix] = (_Float16)eh1[(size_t)(m0 + r) * H_ + cix];
  }
  for (int i = tid; i < BT * DIN; i += 256) {
    const int r = i / DIN, cix = i % DIN;
    Xs[r * LDX + cix] = (_Float16)xg[((size_t)(m0 + r) * T_ + (T_ - 1)) * DIN + cix];
  }
  float b0v[4], b1v[4];
  #pragma unroll
  for (int g = 0; g < 4; ++g) {
    b0v[g] = bs0[g * H_ + nw * 16 + l15];
    b1v[g] = bs1[g * H_ + nw * 16 + l15];
  }
  const float blv = (w < 4) ? bl[(w >> 1) * 16 + l15] : 0.0f;
  v8f c0[2], c1[2];
  #pragma unroll
  for (int mt = 0; mt < 2; ++mt)
    #pragma unroll
    for (int e = 0; e < 8; ++e) {
      const int row = m0 + mt * 16 + e + hi16 * 8;
      const int col = nw * 16 + l15;
      c0[mt][e] = ec0[(size_t)row * H_ + col];
      c1[mt][e] = ec1[(size_t)row * H_ + col];
    }
  __syncthreads();

  for (int t = 0; t < tlen; ++t) {
    // -------- cell 0: [x | h0], K = 160
    v8f acc[2][4];
    #pragma unroll
    for (int mt = 0; mt < 2; ++mt)
      #pragma unroll
      for (int g = 0; g < 4; ++g) acc[mt][g] = splat8(b0v[g]);
    #pragma unroll
    for (int kf = 0; kf < KF0; ++kf) {
      v16h a[2];
      #pragma unroll
      for (int mt = 0; mt < 2; ++mt)
        a[mt] = (kf == 0) ? a_lds(Xs + (mt * 16 + l15) * LDX, 0, hi16)
                          : a_lds(Hs0 + (mt * 16 + l15) * LDH, kf - 1, hi16);
      #pragma unroll
      for (int g = 0; g < 4; ++g) {
        v16h b = b_frag(wf0, (g * 8 + nw) * KF0 + kf, lane);
        #pragma unroll
        for (int mt = 0; mt < 2; ++mt) acc[mt][g] = wmma16(a[mt], b, acc[mt][g]);
      }
    }
    v8f h0n[2];
    #pragma unroll
    for (int mt = 0; mt < 2; ++mt)
      #pragma unroll
      for (int e = 0; e < 8; ++e) {
        float cc = sig_(acc[mt][1][e]) * c0[mt][e] + sig_(acc[mt][0][e]) * tanh_(acc[mt][2][e]);
        c0[mt][e] = cc;
        h0n[mt][e] = sig_(acc[mt][3][e]) * tanh_(cc);
      }
    __syncthreads();
    #pragma unroll
    for (int mt = 0; mt < 2; ++mt)
      #pragma unroll
      for (int e = 0; e < 8; ++e)
        Hs0[(mt * 16 + e + hi16 * 8) * LDH + nw * 16 + l15] = (_Float16)h0n[mt][e];
    __syncthreads();

    // -------- cell 1: [h0 | h1], K = 256 (weights in LDS)
    #pragma unroll
    for (int mt = 0; mt < 2; ++mt)
      #pragma unroll
      for (int g = 0; g < 4; ++g) acc[mt][g] = splat8(b1v[g]);
    #pragma unroll
    for (int kf = 0; kf < KF1; ++kf) {
      const _Float16* src = (kf < 4) ? Hs0 : Hs1;
      const int kl = (kf < 4) ? kf : kf - 4;
      v16h a[2];
      #pragma unroll
      for (int mt = 0; mt < 2; ++mt) a[mt] = a_lds(src + (mt * 16 + l15) * LDH, kl, hi16);
      #pragma unroll
      for (int g = 0; g < 4; ++g) {
        v16h b = b_frag(WL1, (g * 8 + nw) * KF1 + kf, lane);
        #pragma unroll
        for (int mt = 0; mt < 2; ++mt) acc[mt][g] = wmma16(a[mt], b, acc[mt][g]);
      }
    }
    v8f h1n[2];
    #pragma unroll
    for (int mt = 0; mt < 2; ++mt)
      #pragma unroll
      for (int e = 0; e < 8; ++e) {
        float cc = sig_(acc[mt][1][e]) * c1[mt][e] + sig_(acc[mt][0][e]) * tanh_(acc[mt][2][e]);
        c1[mt][e] = cc;
        h1n[mt][e] = sig_(acc[mt][3][e]) * tanh_(cc);
      }
    __syncthreads();
    #pragma unroll
    for (int mt = 0; mt < 2; ++mt)
      #pragma unroll
      for (int e = 0; e < 8; ++e)
        Hs1[(mt * 16 + e + hi16 * 8) * LDH + nw * 16 + l15] = (_Float16)h1n[mt][e];
    __syncthreads();

    // -------- linear: out = h1 @ linW^T + lin_b (waves 0..3, one 16x16 out tile each)
    if (w < 4) {
      const int mt = w & 1, ot = w >> 1;
      v8f accL = splat8(blv);
      #pragma unroll
      for (int kf = 0; kf < KFL; ++kf)
        accL = wmma16(a_lds(Hs1 + (mt * 16 + l15) * LDH, kf, hi16),
                      b_frag(wfl, ot * KFL + kf, lane), accL);
      #pragma unroll
      for (int e = 0; e < 8; ++e) {
        const int m = mt * 16 + e + hi16 * 8;
        const int n = ot * 16 + l15;
        out[((size_t)(m0 + m) * tlen + t) * DIN + n] = accL[e];
        Xs[m * LDX + n] = (_Float16)accL[e];
      }
    }
    __syncthreads();
  }
}

// =====================================================================
extern "C" void kernel_launch(void* const* d_in, const int* in_sizes, int n_in,
                              void* d_out, int out_size, void* d_ws, size_t ws_size,
                              hipStream_t stream) {
  (void)in_sizes; (void)n_in; (void)out_size; (void)ws_size;
  const float* x     = (const float*)d_in[0];
  const int*   tgt   = (const int*)d_in[1];
  const float* eWih0 = (const float*)d_in[2];
  const float* eWhh0 = (const float*)d_in[3];
  const float* ebih0 = (const float*)d_in[4];
  const float* ebhh0 = (const float*)d_in[5];
  const float* eWih1 = (const float*)d_in[6];
  const float* eWhh1 = (const float*)d_in[7];
  const float* ebih1 = (const float*)d_in[8];
  const float* ebhh1 = (const float*)d_in[9];
  const float* dWih0 = (const float*)d_in[10];
  const float* dWhh0 = (const float*)d_in[11];
  const float* dbih0 = (const float*)d_in[12];
  const float* dbhh0 = (const float*)d_in[13];
  const float* dWih1 = (const float*)d_in[14];
  const float* dWhh1 = (const float*)d_in[15];
  const float* dbih1 = (const float*)d_in[16];
  const float* dbhh1 = (const float*)d_in[17];
  const float* linW  = (const float*)d_in[18];
  const float* linb  = (const float*)d_in[19];
  float* out = (float*)d_out;
  char* ws = (char*)d_ws;

  // workspace layout (bytes, 256-aligned)
  constexpr size_t OFF_WE0 = 0;                          // 512*160*2
  constexpr size_t OFF_WE1 = OFF_WE0 + 163840;           // 512*256*2
  constexpr size_t OFF_WD0 = OFF_WE1 + 262144;
  constexpr size_t OFF_WD1 = OFF_WD0 + 163840;
  constexpr size_t OFF_WL  = OFF_WD1 + 262144;           // 32*128*2
  constexpr size_t OFF_BE0 = OFF_WL  + 8192;             // 512*4 each
  constexpr size_t OFF_BE1 = OFF_BE0 + 2048;
  constexpr size_t OFF_BD0 = OFF_BE1 + 2048;
  constexpr size_t OFF_BD1 = OFF_BD0 + 2048;
  constexpr size_t OFF_EH0 = OFF_BD1 + 2048;             // 512*128*4 each
  constexpr size_t OFF_EC0 = OFF_EH0 + 262144;
  constexpr size_t OFF_EH1 = OFF_EC0 + 262144;
  constexpr size_t OFF_EC1 = OFF_EH1 + 262144;

  _Float16* wf_e0 = (_Float16*)(ws + OFF_WE0);
  _Float16* wf_e1 = (_Float16*)(ws + OFF_WE1);
  _Float16* wf_d0 = (_Float16*)(ws + OFF_WD0);
  _Float16* wf_d1 = (_Float16*)(ws + OFF_WD1);
  _Float16* wf_l  = (_Float16*)(ws + OFF_WL);
  float* bs_e0 = (float*)(ws + OFF_BE0);
  float* bs_e1 = (float*)(ws + OFF_BE1);
  float* bs_d0 = (float*)(ws + OFF_BD0);
  float* bs_d1 = (float*)(ws + OFF_BD1);
  float* eh0 = (float*)(ws + OFF_EH0);
  float* ec0 = (float*)(ws + OFF_EC0);
  float* eh1 = (float*)(ws + OFF_EH1);
  float* ec1 = (float*)(ws + OFF_EC1);

  // weight -> f16 fragment layout, bias sums
  prep_wfrag<<<(G4 / 16) * KF0, 32, 0, stream>>>(wf_e0, eWih0, eWhh0, DIN, KF0);
  prep_wfrag<<<(G4 / 16) * KF1, 32, 0, stream>>>(wf_e1, eWih1, eWhh1, H_, KF1);
  prep_wfrag<<<(G4 / 16) * KF0, 32, 0, stream>>>(wf_d0, dWih0, dWhh0, DIN, KF0);
  prep_wfrag<<<(G4 / 16) * KF1, 32, 0, stream>>>(wf_d1, dWih1, dWhh1, H_, KF1);
  prep_wfrag<<<(DIN / 16) * KFL, 32, 0, stream>>>(wf_l, linW, linW, H_, KFL); // Kin==Ktot
  bias_add<<<2, 256, 0, stream>>>(bs_e0, ebih0, ebhh0, G4);
  bias_add<<<2, 256, 0, stream>>>(bs_e1, ebih1, ebhh1, G4);
  bias_add<<<2, 256, 0, stream>>>(bs_d0, dbih0, dbhh0, G4);
  bias_add<<<2, 256, 0, stream>>>(bs_d1, dbih1, dbhh1, G4);

  const size_t smem_enc = (size_t)(G4 * (H_ + H_) + 2 * BT * LDH) * sizeof(_Float16);
  const size_t smem_dec = smem_enc + (size_t)BT * LDX * sizeof(_Float16);
  (void)hipFuncSetAttribute(reinterpret_cast<const void*>(enc_kernel),
                            hipFuncAttributeMaxDynamicSharedMemorySize, (int)smem_enc);
  (void)hipFuncSetAttribute(reinterpret_cast<const void*>(dec_kernel),
                            hipFuncAttributeMaxDynamicSharedMemorySize, (int)smem_dec);

  enc_kernel<<<NWG, 256, smem_enc, stream>>>(x, wf_e0, bs_e0, wf_e1, bs_e1,
                                             eh0, ec0, eh1, ec1);
  dec_kernel<<<NWG, 256, smem_dec, stream>>>(x, tgt, wf_d0, bs_d0, wf_d1, bs_d1,
                                             wf_l, linb, eh0, ec0, eh1, ec1, out);
}
